// SpatialRangeBlock_50251117363406
// MI455X (gfx1250) — matrix operand order repaired and checked by reading, not compiled, not run on hardware
//
#include <hip/hip_runtime.h>

#define DIMC 256
#define HH 64
#define WW 64
#define HWP 4096
#define BB 2
#define NN 49
#define RAD 3
#define LDSP 40   // padded LDS row stride in bf16 elems (80B, 16B aligned, conflict-free)

typedef __attribute__((ext_vector_type(16))) __bf16 v16bf;
typedef __attribute__((ext_vector_type(8)))  __bf16 v8bf;
typedef __attribute__((ext_vector_type(8)))  float  v8f;

__device__ __forceinline__ int reflect_idx(int i, int n) {
  if (i < 0) i = -i;
  if (i >= n) i = 2 * n - 2 - i;
  return i;
}

// ---------------------------------------------------------------------------
// elementwise f32 -> bf16 (weights)
// ---------------------------------------------------------------------------
__global__ __launch_bounds__(256)
void cvt_bf16(const float* __restrict__ in, __bf16* __restrict__ out, int n) {
  int i = blockIdx.x * 256 + threadIdx.x;
  if (i < n) out[i] = (__bf16)in[i];
}

// ---------------------------------------------------------------------------
// tiled transpose+convert: f32 X[C][P] -> bf16 Xt[P][C]  (per batch)
// ---------------------------------------------------------------------------
__global__ __launch_bounds__(64)
void xt_cvt(const float* __restrict__ X, __bf16* __restrict__ Xt) {
  __shared__ float tile[64][65];
  const int p0 = blockIdx.x * 64;
  const int c0 = blockIdx.y * 64;
  const int b  = blockIdx.z;
  const float* Xb = X + (size_t)b * DIMC * HWP;
  __bf16*      Ob = Xt + (size_t)b * DIMC * HWP;
  const int t = threadIdx.x;
#pragma unroll 4
  for (int j = 0; j < 64; ++j) tile[j][t] = Xb[(size_t)(c0 + j) * HWP + p0 + t];
  __syncthreads();
#pragma unroll 4
  for (int i = 0; i < 64; ++i)
    Ob[(size_t)(p0 + i) * DIMC + c0 + t] = (__bf16)tile[t][i];
}

// ---------------------------------------------------------------------------
// GEMM: Y[b] = W(256x256) @ X[b](256x4096) + bias
// bf16 inputs (Xt pixel-major [P][C], Wb row-major [O][I]), f32 accumulate.
// Block = 128 threads (4 waves) -> 64M x 64N tile; each wave: 16M x 64N,
// 4 accumulators, 4 WMMAs per 32-wide K step off one shared A fragment.
// Fragments come straight out of LDS as ds_load_b128 pairs (no cvt in loop).
// ---------------------------------------------------------------------------
__global__ __launch_bounds__(128)
void gemm256_wmma(const __bf16* __restrict__ Xt, const __bf16* __restrict__ Wb,
                  const float* __restrict__ bias, float* __restrict__ Y) {
  __shared__ __bf16 ldsA[64 * LDSP];   // A tile: 64 M-rows x 32 K (padded)
  __shared__ __bf16 ldsB[64 * LDSP];   // B tile: 64 N-rows x 32 K (padded)
  const int nBase = blockIdx.x * 64;
  const int mBase = blockIdx.y * 64;
  const int b     = blockIdx.z;
  const __bf16* Xb = Xt + (size_t)b * DIMC * HWP;
  float*        Yb = Y  + (size_t)b * DIMC * HWP;
  const int t   = threadIdx.x;
  const int l   = t & 31;
  const int wv  = t >> 5;
  const int m16 = l & 15;
  const int hi  = l >> 4;

  v8f acc[4] = {v8f{}, v8f{}, v8f{}, v8f{}};

  for (int k0 = 0; k0 < DIMC; k0 += 32) {
    // cooperative load: 64 rows x 32 K per tile = 256 chunks of 8 bf16 each
#pragma unroll
    for (int i = 0; i < 2; ++i) {
      const int cid = 2 * t + i;
      const int row = cid >> 2, seg = cid & 3;
      *(v8bf*)(ldsA + row * LDSP + seg * 8) =
          *(const v8bf*)(Wb + (size_t)(mBase + row) * DIMC + k0 + seg * 8);
      *(v8bf*)(ldsB + row * LDSP + seg * 8) =
          *(const v8bf*)(Xb + (size_t)(nBase + row) * DIMC + k0 + seg * 8);
    }
    if (k0 + 32 < DIMC) {   // prefetch next K tile (global_prefetch_b8)
      __builtin_prefetch(Wb + (size_t)(mBase + (t >> 1)) * DIMC + k0 + 32, 0, 0);
      __builtin_prefetch(Xb + (size_t)(nBase + (t >> 1)) * DIMC + k0 + 32, 0, 0);
    }
    __syncthreads();

    // A fragment (16-bit A 16x32 wave32 layout): two contiguous 8-elem runs
    const __bf16* arow = ldsA + (size_t)(wv * 16 + m16) * LDSP;
    v8bf alo = *(const v8bf*)(arow + hi * 8);
    v8bf ahi = *(const v8bf*)(arow + 16 + hi * 8);
    v16bf af = __builtin_shufflevector(alo, ahi, 0, 1, 2, 3, 4, 5, 6, 7,
                                       8, 9, 10, 11, 12, 13, 14, 15);
    // B fragments: lane column n, same K order as A: two contiguous 8-elem runs
#pragma unroll
    for (int ns = 0; ns < 4; ++ns) {
      const __bf16* brow = ldsB + (size_t)(ns * 16 + m16) * LDSP;
      v8bf blo = *(const v8bf*)(brow + hi * 8);
      v8bf bhi = *(const v8bf*)(brow + 16 + hi * 8);
      v16bf bf = __builtin_shufflevector(blo, bhi, 0, 1, 2, 3, 4, 5, 6, 7,
                                         8, 9, 10, 11, 12, 13, 14, 15);
      acc[ns] = __builtin_amdgcn_wmma_f32_16x16x32_bf16(
          false, af, false, bf, (short)0, acc[ns], false, false);
    }
    __syncthreads();
  }

#pragma unroll
  for (int ns = 0; ns < 4; ++ns) {
    const int n = nBase + ns * 16 + m16;
#pragma unroll
    for (int r = 0; r < 8; ++r) {   // C/D layout: n=lane%16, m=r+8*(lane/16)
      const int row = mBase + wv * 16 + r + 8 * hi;
      Yb[(size_t)row * HWP + n] = acc[ns][r] + bias[row];
    }
  }
}

// ---------------------------------------------------------------------------
// LayerNorm over channels (256) per pixel, fused f32->bf16 transpose output.
// Reads Y[C][P] coalesced (pixel-per-lane); writes Xt[P][C] coalesced via LDS.
// ---------------------------------------------------------------------------
__global__ __launch_bounds__(64)
void ln_bf16t(const float* __restrict__ Xin, const float* __restrict__ g,
              const float* __restrict__ be, __bf16* __restrict__ Outt,
              int do_silu) {
  __shared__ float tile[64][65];
  const int p0 = blockIdx.x * 64;
  const int b  = blockIdx.y;
  const int t  = threadIdx.x;
  const int p  = p0 + t;
  const float* base = Xin + (size_t)b * DIMC * HWP + p;
  float s = 0.f, s2 = 0.f;
  for (int c = 0; c < DIMC; ++c) {
    float v = base[(size_t)c * HWP];
    s += v; s2 += v * v;
  }
  const float mean = s * (1.f / DIMC);
  const float var  = s2 * (1.f / DIMC) - mean * mean;
  const float rstd = rsqrtf(var + 1e-6f);
  __bf16* Ob = Outt + (size_t)b * DIMC * HWP;
  for (int chunk = 0; chunk < 4; ++chunk) {
#pragma unroll 4
    for (int j = 0; j < 64; ++j) {
      const int c = chunk * 64 + j;
      float y = g[c] * (base[(size_t)c * HWP] - mean) * rstd + be[c];
      if (do_silu) y = y / (1.f + __expf(-y));
      tile[j][t] = y;
    }
    __syncthreads();
#pragma unroll 4
    for (int i = 0; i < 64; ++i)
      Ob[(size_t)(p0 + i) * DIMC + chunk * 64 + t] = (__bf16)tile[t][i];
    __syncthreads();
  }
}

// ---------------------------------------------------------------------------
// sim[pix][n] = <px(center), px(neigh_n)> / 16   (reflect pad), one pixel/block
// ---------------------------------------------------------------------------
__global__ __launch_bounds__(64)
void sim_kernel(const float* __restrict__ px, float* __restrict__ sim) {
  __shared__ float ctr[DIMC];
  const int pix = blockIdx.x;
  const int b = pix / HWP, p = pix % HWP;
  const int h = p >> 6, w = p & 63;
  const float* pb = px + (size_t)b * DIMC * HWP;
  for (int c = threadIdx.x; c < DIMC; c += 64) ctr[c] = pb[(size_t)c * HWP + p];
  __syncthreads();
  const int t = threadIdx.x;
  if (t < NN) {
    int dy = t / 7 - RAD, dx = t % 7 - RAD;
    int np = reflect_idx(h + dy, HH) * WW + reflect_idx(w + dx, WW);
    float s = 0.f;
    for (int c = 0; c < DIMC; ++c) s += ctr[c] * pb[(size_t)c * HWP + np];
    sim[(size_t)pix * NN + t] = s * (1.f / 16.f);   // / sqrt(256)
  }
}

// ---------------------------------------------------------------------------
// softmax over 49 * gaussian spatial kernel -> ck0
// ---------------------------------------------------------------------------
__global__ __launch_bounds__(64)
void softmax_gauss(const float* __restrict__ sim, const float* __restrict__ sigma,
                   float* __restrict__ ck0) {
  __shared__ float sv[64];
  const int pix = blockIdx.x;
  const int t = threadIdx.x;
  float val = (t < NN) ? sim[(size_t)pix * NN + t] : -3.4e38f;
  sv[t] = val; __syncthreads();
  for (int o = 32; o >= 1; o >>= 1) { if (t < o) sv[t] = fmaxf(sv[t], sv[t + o]); __syncthreads(); }
  float mx = sv[0]; __syncthreads();
  float e = (t < NN) ? __expf(val - mx) : 0.f;
  sv[t] = e; __syncthreads();
  for (int o = 32; o >= 1; o >>= 1) { if (t < o) sv[t] += sv[t + o]; __syncthreads(); }
  float inv = 1.f / sv[0];
  if (t < NN) {
    float sg = sigma[0];
    float x = -1.f + (2.f / 6.f) * (float)(t / 7);
    float y = -1.f + (2.f / 6.f) * (float)(t % 7);
    float gauss = __expf(-(x * x + y * y) / (2.f * sg * sg));
    ck0[(size_t)pix * NN + t] = e * inv * gauss;
  }
}

// ---------------------------------------------------------------------------
// fixup conv1: fx1 = fp_w1 @ concat(ck0, semantic) + b1   (305 -> 49)
// ---------------------------------------------------------------------------
__global__ __launch_bounds__(64)
void fixup1(const float* __restrict__ ck0, const float* __restrict__ sem,
            const float* __restrict__ w1, const float* __restrict__ b1,
            float* __restrict__ fx1) {
  __shared__ float lck[NN];
  __shared__ float lsem[DIMC];
  const int pix = blockIdx.x;
  const int b = pix / HWP, p = pix % HWP;
  const int t = threadIdx.x;
  if (t < NN) lck[t] = ck0[(size_t)pix * NN + t];
  const float* sb = sem + (size_t)b * DIMC * HWP + p;
  for (int c = t; c < DIMC; c += 64) lsem[c] = sb[(size_t)c * HWP];
  __syncthreads();
  if (t < NN) {
    const float* wr = w1 + (size_t)t * (NN + DIMC);
    float s = b1[t];
#pragma unroll 7
    for (int i = 0; i < NN; ++i) s += wr[i] * lck[i];
#pragma unroll 8
    for (int c = 0; c < DIMC; ++c) s += wr[NN + c] * lsem[c];
    fx1[(size_t)pix * NN + t] = s;
  }
}

// ---------------------------------------------------------------------------
// LN over 49 channels + SiLU, one wave per pixel
// ---------------------------------------------------------------------------
__global__ __launch_bounds__(32)
void ln49_silu(const float* __restrict__ Xin, const float* __restrict__ g,
               const float* __restrict__ be, float* __restrict__ Yout) {
  const int pix = blockIdx.x;
  const int t = threadIdx.x, t2 = t + 32;
  float v0 = (t  < NN) ? Xin[(size_t)pix * NN + t]  : 0.f;
  float v1 = (t2 < NN) ? Xin[(size_t)pix * NN + t2] : 0.f;
  float s = v0 + v1;
  for (int o = 16; o >= 1; o >>= 1) s += __shfl_xor(s, o, 32);
  float mean = s * (1.f / NN);
  float d0 = (t  < NN) ? v0 - mean : 0.f;
  float d1 = (t2 < NN) ? v1 - mean : 0.f;
  float var = d0 * d0 + d1 * d1;
  for (int o = 16; o >= 1; o >>= 1) var += __shfl_xor(var, o, 32);
  float rstd = rsqrtf(var * (1.f / NN) + 1e-6f);
  if (t < NN) {
    float y = g[t] * d0 * rstd + be[t];
    Yout[(size_t)pix * NN + t] = y / (1.f + __expf(-y));
  }
  if (t2 < NN) {
    float y = g[t2] * d1 * rstd + be[t2];
    Yout[(size_t)pix * NN + t2] = y / (1.f + __expf(-y));
  }
}

// ---------------------------------------------------------------------------
// fixup conv2 + sigmoid gate + renormalize over 49 -> final ck
// ---------------------------------------------------------------------------
__global__ __launch_bounds__(64)
void fixup2_norm(const float* __restrict__ hbuf, const float* __restrict__ ck0,
                 const float* __restrict__ w2, const float* __restrict__ b2,
                 float* __restrict__ ck) {
  __shared__ float lh[NN];
  __shared__ float red[64];
  const int pix = blockIdx.x;
  const int t = threadIdx.x;
  if (t < NN) lh[t] = hbuf[(size_t)pix * NN + t];
  __syncthreads();
  float cval = 0.f;
  if (t < NN) {
    const float* wr = w2 + (size_t)t * NN;
    float s = b2[t];
#pragma unroll 7
    for (int i = 0; i < NN; ++i) s += wr[i] * lh[i];
    float gate = 1.f / (1.f + __expf(-s));
    cval = ck0[(size_t)pix * NN + t] * (1.f + gate);
  }
  red[t] = cval; __syncthreads();
  for (int o = 32; o >= 1; o >>= 1) { if (t < o) red[t] += red[t + o]; __syncthreads(); }
  float inv = 1.f / (red[0] + 1e-7f);
  if (t < NN) ck[(size_t)pix * NN + t] = cval * inv;
}

// ---------------------------------------------------------------------------
// out[b,c,p] = sum_n ck[b,p,n] * spatial[b,c,neigh_n(p)]; one (b,c,h) row/block
// ---------------------------------------------------------------------------
__global__ __launch_bounds__(64)
void aggregate(const float* __restrict__ spat, const float* __restrict__ ck,
               float* __restrict__ outb) {
  const int id = blockIdx.x;                 // b*DIMC*HH + c*HH + h
  const int h = id % HH;
  const int c = (id / HH) % DIMC;
  const int b = id / (HH * DIMC);
  const int w = threadIdx.x;
  const float* sp  = spat + (size_t)b * DIMC * HWP + (size_t)c * HWP;
  const float* ckp = ck + ((size_t)b * HWP + h * WW + w) * NN;
  float acc = 0.f;
#pragma unroll
  for (int n = 0; n < NN; ++n) {
    int hy = reflect_idx(h + n / 7 - RAD, HH);
    int wx = reflect_idx(w + n % 7 - RAD, WW);
    acc += ckp[n] * sp[hy * WW + wx];
  }
  outb[(size_t)b * DIMC * HWP + (size_t)c * HWP + h * WW + w] = acc;
}

// ---------------------------------------------------------------------------
extern "C" void kernel_launch(void* const* d_in, const int* in_sizes, int n_in,
                              void* d_out, int out_size, void* d_ws, size_t ws_size,
                              hipStream_t stream) {
  const float* spatial  = (const float*)d_in[0];
  const float* semantic = (const float*)d_in[1];
  const float* rp_w1 = (const float*)d_in[2];
  const float* rp_b1 = (const float*)d_in[3];
  const float* rp_g  = (const float*)d_in[4];
  const float* rp_be = (const float*)d_in[5];
  const float* rp_w2 = (const float*)d_in[6];
  const float* rp_b2 = (const float*)d_in[7];
  const float* fp_w1 = (const float*)d_in[8];
  const float* fp_b1 = (const float*)d_in[9];
  const float* fp_g  = (const float*)d_in[10];
  const float* fp_be = (const float*)d_in[11];
  const float* fp_w2 = (const float*)d_in[12];
  const float* fp_b2 = (const float*)d_in[13];
  const float* op_w1 = (const float*)d_in[14];
  const float* op_b1 = (const float*)d_in[15];
  const float* op_g  = (const float*)d_in[16];
  const float* op_be = (const float*)d_in[17];
  const float* op_w2 = (const float*)d_in[18];
  const float* op_b2 = (const float*)d_in[19];
  const float* sigma = (const float*)d_in[20];

  char* ws = (char*)d_ws;
  const size_t SZF = (size_t)BB * DIMC * HWP * sizeof(float);   // 8 MB
  const size_t SZS = (size_t)BB * HWP * NN * sizeof(float);     // ~1.6 MB
  const size_t SZT = (size_t)BB * DIMC * HWP * sizeof(__bf16);  // 4 MB
  const size_t SZW = (size_t)DIMC * DIMC * sizeof(__bf16);      // 128 KB
  float*  buf0 = (float*)(ws);
  float*  buf1 = (float*)(ws + SZF);
  float*  buf2 = (float*)(ws + 2 * SZF);
  float*  simA = (float*)(ws + 3 * SZF);
  float*  simB = (float*)(ws + 3 * SZF + SZS);
  float*  fxA  = (float*)(ws + 3 * SZF + 2 * SZS);
  float*  fxB  = (float*)(ws + 3 * SZF + 3 * SZS);
  __bf16* bfT  = (__bf16*)(ws + 3 * SZF + 4 * SZS);
  __bf16* wbA  = (__bf16*)(ws + 3 * SZF + 4 * SZS + SZT);
  __bf16* wbB  = (__bf16*)(ws + 3 * SZF + 4 * SZS + SZT + SZW);
  __bf16* wbC  = (__bf16*)(ws + 3 * SZF + 4 * SZS + SZT + 2 * SZW);
  __bf16* wbD  = (__bf16*)(ws + 3 * SZF + 4 * SZS + SZT + 3 * SZW);

  dim3 gg(HWP / 64, DIMC / 64, BB);   // GEMM: 64x64 tiles, 128 threads
  dim3 gt(HWP / 64, DIMC / 64, BB);   // transpose-cvt tiles
  dim3 gl(HWP / 64, BB);              // LN: 64 pixels per block
  const int NPIX = BB * HWP;
  const int NW = DIMC * DIMC;

  // one-time weight conversion to bf16
  cvt_bf16<<<NW / 256, 256, 0, stream>>>(rp_w1, wbA, NW);
  cvt_bf16<<<NW / 256, 256, 0, stream>>>(rp_w2, wbB, NW);
  cvt_bf16<<<NW / 256, 256, 0, stream>>>(op_w1, wbC, NW);
  cvt_bf16<<<NW / 256, 256, 0, stream>>>(op_w2, wbD, NW);

  // range_proj
  xt_cvt<<<gt, 64, 0, stream>>>(semantic, bfT);
  gemm256_wmma<<<gg, 128, 0, stream>>>(bfT, wbA, rp_b1, buf0);
  ln_bf16t<<<gl, 64, 0, stream>>>(buf0, rp_g, rp_be, bfT, 1);
  gemm256_wmma<<<gg, 128, 0, stream>>>(bfT, wbB, rp_b2, buf2);     // px (f32)

  // range kernel + gaussian
  sim_kernel<<<NPIX, 64, 0, stream>>>(buf2, simA);
  softmax_gauss<<<NPIX, 64, 0, stream>>>(simA, sigma, simB);       // ck0

  // gated fixup
  fixup1<<<NPIX, 64, 0, stream>>>(simB, semantic, fp_w1, fp_b1, fxA);
  ln49_silu<<<NPIX, 32, 0, stream>>>(fxA, fp_g, fp_be, fxB);
  fixup2_norm<<<NPIX, 64, 0, stream>>>(fxB, simB, fp_w2, fp_b2, simA);  // ck

  // weighted neighborhood sum
  aggregate<<<BB * DIMC * HH, 64, 0, stream>>>(spatial, simA, buf0);

  // output_proj
  xt_cvt<<<gt, 64, 0, stream>>>(buf0, bfT);
  gemm256_wmma<<<gg, 128, 0, stream>>>(bfT, wbC, op_b1, buf1);
  ln_bf16t<<<gl, 64, 0, stream>>>(buf1, op_g, op_be, bfT, 0);
  gemm256_wmma<<<gg, 128, 0, stream>>>(bfT, wbD, op_b2, (float*)d_out);
}
